// Decoder_11948599018102
// MI455X (gfx1250) — matrix operand order repaired and checked by reading, not compiled, not run on hardware
//
#include <hip/hip_runtime.h>
#include <hip/hip_bf16.h>
#include <cstdint>

// Problem dims
#define B_    64
#define P_    196
#define T_    64
#define V_    10000
#define E_    512
#define D_    512
#define ENC_  512
#define TD    63              // decode steps = T-1
#define G4    2048            // 4*D
#define KC    1536            // E + ENC + D (fused gate GEMM K)
#define MFC   4032            // TD*B rows for the big FC GEMM

// Async global->LDS path (gfx1250): guarded so the file compiles on both
// toolchains; fallback is global load + ds_store (same barrier protocol).
#if defined(__has_builtin)
#  if __has_builtin(__builtin_amdgcn_global_load_async_to_lds_b128) && \
      __has_builtin(__builtin_amdgcn_s_wait_asynccnt)
#    define HAVE_ASYNC_LDS 1
#  endif
#endif
#ifndef HAVE_ASYNC_LDS
#  define HAVE_ASYNC_LDS 0
#endif

typedef __attribute__((ext_vector_type(16))) __bf16 v16bf;
typedef __attribute__((ext_vector_type(8)))  float  v8f;
typedef __attribute__((ext_vector_type(4)))  int    v4i;

#define ASG __attribute__((address_space(1)))
#define ASL __attribute__((address_space(3)))

union ABFrag { uint4 q[2]; v16bf v; };

// float -> bf16 (round-to-nearest-even), stored as raw u16
static __device__ __forceinline__ unsigned short f2bf(float f) {
    unsigned u = __float_as_uint(f);
    u += 0x7FFFu + ((u >> 16) & 1u);
    return (unsigned short)(u >> 16);
}

// 16B global -> LDS copy: async on gfx1250 (ASYNCcnt-tracked), sync fallback.
static __device__ __forceinline__ void cp16_g2l(const unsigned short* src,
                                                unsigned short* dst_lds) {
#if HAVE_ASYNC_LDS
    __builtin_amdgcn_global_load_async_to_lds_b128(
        (ASG v4i*)src, (ASL v4i*)dst_lds, 0, 0);
#else
    *(uint4*)dst_lds = *(const uint4*)src;
#endif
}

static __device__ __forceinline__ void wait_stage() {
#if HAVE_ASYNC_LDS
    __builtin_amdgcn_s_wait_asynccnt(0);
#endif
}

// A fragment: 16x32 bf16 tile. lane<16: M=lane, K = k0..k0+15
//             lane>=16: M=lane-16, K = k0+16..k0+31
static __device__ __forceinline__ v16bf load_frag_a(const unsigned short* base,
                                                    int ldk, int row, int k0, int lane) {
    int khalf = (lane < 16) ? 0 : 16;
    const unsigned short* p = base + row * ldk + k0 + khalf;
    ABFrag u;
    u.q[0] = *(const uint4*)(p);
    u.q[1] = *(const uint4*)(p + 8);
    return u.v;
}

// B fragment from row-major W[n,k] in global memory.
// lane<16: N=lane, K = k0..k0+15 ; lane>=16: N=lane-16, K = k0+16..k0+31
static __device__ __forceinline__ v16bf load_frag_b(const unsigned short* base,
                                                    int ldk, int nrow, int k0, int lane) {
    int kofs = (lane < 16) ? 0 : 16;
    const unsigned short* p = base + nrow * ldk + k0 + kofs;
    ABFrag u;
    u.q[0] = *(const uint4*)(p);
    u.q[1] = *(const uint4*)(p + 8);
    return u.v;
}

#define LROW 40   // LDS B-tile row stride in shorts (80B: 16B-aligned, banks staggered)

// B fragment from an LDS-staged 64x32 tile (row-major, LROW stride)
static __device__ __forceinline__ v16bf load_frag_b_lds(const unsigned short* lbuf,
                                                        int nloc, int lane) {
    int kofs = (lane < 16) ? 0 : 16;
    const unsigned short* p = lbuf + nloc * LROW + kofs;
    ABFrag u;
    u.q[0] = *(const uint4*)(p);
    u.q[1] = *(const uint4*)(p + 8);
    return u.v;
}

static __device__ __forceinline__ v8f splat8(float v) {
    v8f z;
#pragma unroll
    for (int r = 0; r < 8; ++r) z[r] = v;
    return z;
}

// ---------------- setup kernels ----------------

__global__ __launch_bounds__(256) void k_sumenc(const float* __restrict__ enc,
                                                float* sum_f, float* mean_f,
                                                unsigned short* sum_bf) {
    int idx = blockIdx.x * blockDim.x + threadIdx.x;   // B_*ENC_
    int b = idx >> 9, e = idx & 511;
    float s = 0.f;
    const float* p = enc + (size_t)(b * P_) * ENC_ + e;
    for (int pp = 0; pp < P_; ++pp) s += p[(size_t)pp * ENC_];
    sum_f[idx]  = s;
    mean_f[idx] = s * (1.0f / (float)P_);
    sum_bf[idx] = f2bf(s);
}

__global__ __launch_bounds__(256) void k_h0c0(const float* __restrict__ mean_f,
                                              const float* __restrict__ Wh0, const float* __restrict__ bh0,
                                              const float* __restrict__ Wc0, const float* __restrict__ bc0,
                                              float* h, float* c, unsigned short* h_bf) {
    int idx = blockIdx.x * blockDim.x + threadIdx.x;   // B_*D_
    int b = idx >> 9, d = idx & 511;
    float sh = bh0[d], sc = bc0[d];
    const float* m  = mean_f + b * ENC_;
    const float* wh = Wh0 + (size_t)d * ENC_;
    const float* wc = Wc0 + (size_t)d * ENC_;
    for (int k = 0; k < ENC_; ++k) { float mv = m[k]; sh = fmaf(mv, wh[k], sh); sc = fmaf(mv, wc[k], sc); }
    h[idx] = sh; c[idx] = sc; h_bf[idx] = f2bf(sh);
}

__global__ __launch_bounds__(256) void k_pack_wcat(const float* __restrict__ Wih,
                                                   const float* __restrict__ Whh,
                                                   unsigned short* Wcat) {
    int idx = blockIdx.x * blockDim.x + threadIdx.x;   // G4*KC
    if (idx >= G4 * KC) return;
    int n = idx / KC, k = idx - n * KC;
    float v = (k < (E_ + ENC_)) ? Wih[(size_t)n * (E_ + ENC_) + k]
                                : Whh[(size_t)n * D_ + (k - (E_ + ENC_))];
    Wcat[idx] = f2bf(v);
}

__global__ __launch_bounds__(256) void k_conv_wfc(const float* __restrict__ Wfc,
                                                  unsigned short* Wfc_bf) {
    int idx = blockIdx.x * blockDim.x + threadIdx.x;   // V_*D_
    if (idx >= V_ * D_) return;
    Wfc_bf[idx] = f2bf(Wfc[idx]);
}

__global__ __launch_bounds__(256) void k_gather_emb(const float* __restrict__ emb,
                                                    const int* __restrict__ caps,
                                                    unsigned short* Aemb) {
    int idx = blockIdx.x * blockDim.x + threadIdx.x;   // MFC*E_
    if (idx >= MFC * E_) return;
    int r = idx >> 9, d = idx & 511;
    int t = r >> 6, b = r & 63;
    int tok = caps[b * T_ + t];
    Aemb[idx] = f2bf(emb[(size_t)tok * E_ + d]);
}

__global__ __launch_bounds__(256) void k_biassum(const float* __restrict__ bih,
                                                 const float* __restrict__ bhh,
                                                 float* biassum) {
    int idx = blockIdx.x * blockDim.x + threadIdx.x;
    if (idx < G4) biassum[idx] = bih[idx] + bhh[idx];
}

// ---------------- per-step fused gate GEMM ----------------
// gates[64 x 2048] = [emb_t | sum_enc | h] (64 x 1536, bf16) @ Wcat^T + (b_ih+b_hh)
// Two-phase manual modulo schedule: disjoint fragment sets alternate roles,
// so loads run one phase ahead of their WMMAs with zero register copies.
struct Acc4 { v8f a0, a1, a2, a3; };

static __device__ __forceinline__ void gemm_seg(const unsigned short* __restrict__ abase,
                                                const unsigned short* __restrict__ Wcat,
                                                int wcol_base, int arow, int n0,
                                                int col, int lane, Acc4& acc) {
    // set 0 fragments (even k-steps), set 1 fragments (odd k-steps)
    v16bf a0, p0, q0, r0, s0;
    v16bf a1, p1, q1, r1, s1;

    a0 = load_frag_a(abase, 512, arow, 0, lane);
    p0 = load_frag_b(Wcat, KC, n0 +  0 + col, wcol_base, lane);
    q0 = load_frag_b(Wcat, KC, n0 + 16 + col, wcol_base, lane);
    r0 = load_frag_b(Wcat, KC, n0 + 32 + col, wcol_base, lane);
    s0 = load_frag_b(Wcat, KC, n0 + 48 + col, wcol_base, lane);

    for (int kk = 0; kk < ENC_ / 32; kk += 2) {
        // phase A: load set1 for kk+1, compute kk with set0
        {
            int k1 = wcol_base + (kk + 1) * 32;
            a1 = load_frag_a(abase, 512, arow, (kk + 1) * 32, lane);
            p1 = load_frag_b(Wcat, KC, n0 +  0 + col, k1, lane);
            q1 = load_frag_b(Wcat, KC, n0 + 16 + col, k1, lane);
            r1 = load_frag_b(Wcat, KC, n0 + 32 + col, k1, lane);
            s1 = load_frag_b(Wcat, KC, n0 + 48 + col, k1, lane);
        }
        acc.a0 = __builtin_amdgcn_wmma_f32_16x16x32_bf16(false, a0, false, p0, (short)0, acc.a0, false, false);
        acc.a1 = __builtin_amdgcn_wmma_f32_16x16x32_bf16(false, a0, false, q0, (short)0, acc.a1, false, false);
        acc.a2 = __builtin_amdgcn_wmma_f32_16x16x32_bf16(false, a0, false, r0, (short)0, acc.a2, false, false);
        acc.a3 = __builtin_amdgcn_wmma_f32_16x16x32_bf16(false, a0, false, s0, (short)0, acc.a3, false, false);

        // phase B: load set0 for kk+2, compute kk+1 with set1
        if (kk + 2 < ENC_ / 32) {
            int k2 = wcol_base + (kk + 2) * 32;
            a0 = load_frag_a(abase, 512, arow, (kk + 2) * 32, lane);
            p0 = load_frag_b(Wcat, KC, n0 +  0 + col, k2, lane);
            q0 = load_frag_b(Wcat, KC, n0 + 16 + col, k2, lane);
            r0 = load_frag_b(Wcat, KC, n0 + 32 + col, k2, lane);
            s0 = load_frag_b(Wcat, KC, n0 + 48 + col, k2, lane);
        }
        acc.a0 = __builtin_amdgcn_wmma_f32_16x16x32_bf16(false, a1, false, p1, (short)0, acc.a0, false, false);
        acc.a1 = __builtin_amdgcn_wmma_f32_16x16x32_bf16(false, a1, false, q1, (short)0, acc.a1, false, false);
        acc.a2 = __builtin_amdgcn_wmma_f32_16x16x32_bf16(false, a1, false, r1, (short)0, acc.a2, false, false);
        acc.a3 = __builtin_amdgcn_wmma_f32_16x16x32_bf16(false, a1, false, s1, (short)0, acc.a3, false, false);
    }
}

__global__ __launch_bounds__(256) void k_gates(const unsigned short* __restrict__ Aemb,
                                               const unsigned short* __restrict__ senc_bf,
                                               const unsigned short* __restrict__ h_bf,
                                               const unsigned short* __restrict__ Wcat,
                                               const float* __restrict__ biassum,
                                               float* __restrict__ gates, int t) {
    int wid  = (blockIdx.x * blockDim.x + threadIdx.x) >> 5;   // 0..127
    int lane = threadIdx.x & 31;
    int mtile = wid & 3, ntile = wid >> 2;
    int m0 = mtile * 16, n0 = ntile * 64;
    int col = lane & 15;
    int arow = m0 + col;

    Acc4 acc;
    acc.a0 = splat8(biassum[n0 +  0 + col]);
    acc.a1 = splat8(biassum[n0 + 16 + col]);
    acc.a2 = splat8(biassum[n0 + 32 + col]);
    acc.a3 = splat8(biassum[n0 + 48 + col]);

    const unsigned short* aemb_t = Aemb + (size_t)(t * B_) * E_;
    gemm_seg(aemb_t,  Wcat,           0, arow, n0, col, lane, acc);   // emb_t
    gemm_seg(senc_bf, Wcat,        ENC_, arow, n0, col, lane, acc);   // sum_enc
    gemm_seg(h_bf,    Wcat, ENC_ + ENC_, arow, n0, col, lane, acc);   // h

    int rbase = m0 + ((lane >> 4) << 3);
    v8f av[4] = { acc.a0, acc.a1, acc.a2, acc.a3 };
#pragma unroll
    for (int j = 0; j < 4; ++j) {
        int n = n0 + 16 * j + col;
#pragma unroll
        for (int r = 0; r < 8; ++r)
            gates[(size_t)(rbase + r) * G4 + n] = av[j][r];
    }
}

// ---------------- per-step pointwise LSTM ----------------
__global__ __launch_bounds__(256) void k_lstm(const float* __restrict__ gates,
                                              float* __restrict__ h, float* __restrict__ c,
                                              unsigned short* __restrict__ h_bf,
                                              unsigned short* __restrict__ hall_bf,
                                              const int* __restrict__ caplen, int t) {
    int idx = blockIdx.x * blockDim.x + threadIdx.x;   // B_*D_
    int b = idx >> 9, d = idx & 511;
    const float* g = gates + (size_t)b * G4;
    float gi = g[d], gf = g[D_ + d], gg = g[2 * D_ + d], go = g[3 * D_ + d];
    float i = 1.f / (1.f + __expf(-gi));
    float f = 1.f / (1.f + __expf(-gf));
    float o = 1.f / (1.f + __expf(-go));
    float gt = tanhf(gg);
    float cn = f * c[idx] + i * gt;
    float hn = o * tanhf(cn);
    // h_new always feeds the vocab projection (reference uses h_new, masked at output)
    hall_bf[(size_t)((t << 6) + b) * D_ + d] = f2bf(hn);
    // recurrent state only advances while active
    if (t < caplen[b] - 1) {
        h[idx] = hn; c[idx] = cn; h_bf[idx] = f2bf(hn);
    }
}

// ---------------- batched vocab projection ----------------
// out[(b*63+t)*V + n] = active ? Hall[t*64+b] . Wfc[n] + bfc[n] : 0
// Block = 8 waves sharing one 64-col ntile. The 64x32 bf16 B-tile (4KB) is
// staged per k-step into double-buffered LDS (async global->LDS on gfx1250),
// then all waves read B via ds_load. Two-phase schedule: A-fragment sets and
// LDS buffers alternate between phases, no register copies.
static __device__ __forceinline__ void fc_compute(const unsigned short* lb,
                                                  const v16bf (&a)[4],
                                                  v8f (&acc)[4][4],
                                                  int col, int lane) {
#pragma unroll
    for (int j = 0; j < 4; ++j) {
        v16bf b = load_frag_b_lds(lb, 16 * j + col, lane);
#pragma unroll
        for (int i = 0; i < 4; ++i)
            acc[i][j] = __builtin_amdgcn_wmma_f32_16x16x32_bf16(
                false, a[i], false, b, (short)0, acc[i][j], false, false);
    }
}

__global__ __launch_bounds__(256) void k_fc(const unsigned short* __restrict__ Hall,
                                            const unsigned short* __restrict__ Wfc,
                                            const float* __restrict__ bfc,
                                            const int* __restrict__ caplen,
                                            float* __restrict__ out) {
    __shared__ unsigned short lbuf[2][64 * LROW];        // 2 x 5120B

    const int MT = MFC / 64;                 // 63
    const int NK = D_ / 32;                  // 16 k-steps
    int tid  = threadIdx.x;
    int lane = tid & 31;
    int wave = tid >> 5;
    int ntile = blockIdx.x;
    int mtile = blockIdx.y * 8 + wave;
    bool active = (mtile < MT);              // uniform per wave
    int m0 = mtile * 64, n0 = ntile * 64;
    int col = lane & 15;

    // staging assignment: thread -> (row, 16B chunk) of the 64x32 B-tile
    int srow = tid >> 2, schunk = (tid & 3) * 8;
    int snr = n0 + srow; if (snr >= V_) snr = V_ - 1;    // clamp OOB rows
    const unsigned short* sbase = Wfc + (size_t)snr * D_ + schunk;
    unsigned short* ldst0 = &lbuf[0][srow * LROW + schunk];
    unsigned short* ldst1 = &lbuf[1][srow * LROW + schunk];

    v8f acc[4][4];
#pragma unroll
    for (int j = 0; j < 4; ++j) {
        int n = n0 + 16 * j + col;
        float bv = (active && n < V_) ? bfc[n] : 0.f;
#pragma unroll
        for (int i = 0; i < 4; ++i) acc[i][j] = splat8(bv);
    }

    // stage k-step 0 into buffer 0; preload A set0
    cp16_g2l(sbase, ldst0);
    v16bf aset0[4], aset1[4];
    if (active) {
#pragma unroll
        for (int i = 0; i < 4; ++i)
            aset0[i] = load_frag_a(Hall, D_, m0 + 16 * i + col, 0, lane);
    }

    for (int kk = 0; kk < NK; kk += 2) {
        // ---- phase A: buffer0/set0 compute kk; stage kk+1 -> buffer1 ----
        wait_stage();
        __syncthreads();                     // stage(kk) visible; compute(kk-1) done
        cp16_g2l(sbase + (kk + 1) * 32, ldst1);          // kk+1 < NK always (NK even)
        if (active) {
#pragma unroll
            for (int i = 0; i < 4; ++i)
                aset1[i] = load_frag_a(Hall, D_, m0 + 16 * i + col, (kk + 1) * 32, lane);
            fc_compute(lbuf[0], aset0, acc, col, lane);
        }
        // ---- phase B: buffer1/set1 compute kk+1; stage kk+2 -> buffer0 ----
        wait_stage();
        __syncthreads();                     // stage(kk+1) visible; compute(kk) done
        if (kk + 2 < NK) {
            cp16_g2l(sbase + (kk + 2) * 32, ldst0);
            if (active) {
#pragma unroll
                for (int i = 0; i < 4; ++i)
                    aset0[i] = load_frag_a(Hall, D_, m0 + 16 * i + col, (kk + 2) * 32, lane);
            }
        }
        if (active) fc_compute(lbuf[1], aset1, acc, col, lane);
    }

    if (active) {
        int rsub = (lane >> 4) << 3;
#pragma unroll
        for (int j = 0; j < 4; ++j) {
            int n = n0 + 16 * j + col;
            if (n < V_) {
#pragma unroll
                for (int i = 0; i < 4; ++i) {
#pragma unroll
                    for (int r = 0; r < 8; ++r) {
                        int m = m0 + 16 * i + rsub + r;
                        int tt = m >> 6, bb = m & 63;
                        bool act = tt < (caplen[bb] - 1);
                        out[(size_t)(bb * TD + tt) * V_ + n] = act ? acc[i][j][r] : 0.f;
                    }
                }
            }
        }
    }
}

// ---------------- host launcher ----------------
extern "C" void kernel_launch(void* const* d_in, const int* in_sizes, int n_in,
                              void* d_out, int out_size, void* d_ws, size_t ws_size,
                              hipStream_t stream) {
    const float* enc   = (const float*)d_in[0];
    const int*   caps  = (const int*)  d_in[1];
    const int*   clen  = (const int*)  d_in[2];
    const float* emb   = (const float*)d_in[3];
    const float* Wih   = (const float*)d_in[4];
    const float* bih   = (const float*)d_in[5];
    const float* Whh   = (const float*)d_in[6];
    const float* bhh   = (const float*)d_in[7];
    const float* Wh0   = (const float*)d_in[8];
    const float* bh0   = (const float*)d_in[9];
    const float* Wc0   = (const float*)d_in[10];
    const float* bc0   = (const float*)d_in[11];
    const float* Wfc   = (const float*)d_in[12];
    const float* bfc   = (const float*)d_in[13];
    float* out = (float*)d_out;

    char* ws = (char*)d_ws;
    size_t off = 0;
    auto alloc = [&](size_t bytes) { char* p = ws + off; off += (bytes + 255) & ~size_t(255); return p; };
    float*          sum_f   = (float*)         alloc(B_ * ENC_ * 4);
    float*          mean_f  = (float*)         alloc(B_ * ENC_ * 4);
    unsigned short* sum_bf  = (unsigned short*)alloc(B_ * ENC_ * 2);
    float*          h_f     = (float*)         alloc(B_ * D_ * 4);
    float*          c_f     = (float*)         alloc(B_ * D_ * 4);
    unsigned short* h_bf    = (unsigned short*)alloc(B_ * D_ * 2);
    float*          biassum = (float*)         alloc(G4 * 4);
    float*          gates   = (float*)         alloc(B_ * G4 * 4);
    unsigned short* Wcat    = (unsigned short*)alloc((size_t)G4 * KC * 2);
    unsigned short* Wfc_bf  = (unsigned short*)alloc((size_t)V_ * D_ * 2);
    unsigned short* Aemb    = (unsigned short*)alloc((size_t)MFC * E_ * 2);
    unsigned short* Hall    = (unsigned short*)alloc((size_t)MFC * D_ * 2);

    // setup
    k_sumenc    <<<(B_ * ENC_) / 256, 256, 0, stream>>>(enc, sum_f, mean_f, sum_bf);
    k_h0c0      <<<(B_ * D_) / 256, 256, 0, stream>>>(mean_f, Wh0, bh0, Wc0, bc0, h_f, c_f, h_bf);
    k_pack_wcat <<<(G4 * KC + 255) / 256, 256, 0, stream>>>(Wih, Whh, Wcat);
    k_conv_wfc  <<<(V_ * D_ + 255) / 256, 256, 0, stream>>>(Wfc, Wfc_bf);
    k_gather_emb<<<(MFC * E_ + 255) / 256, 256, 0, stream>>>(emb, caps, Aemb);
    k_biassum   <<<(G4 + 255) / 256, 256, 0, stream>>>(bih, bhh, biassum);

    // sequential recurrence: fused K=1536 gate GEMM + pointwise per step
    for (int t = 0; t < TD; ++t) {
        k_gates<<<16, 256, 0, stream>>>(Aemb, sum_bf, h_bf, Wcat, biassum, gates, t);
        k_lstm <<<(B_ * D_) / 256, 256, 0, stream>>>(gates, h_f, c_f, h_bf, Hall, clen, t);
    }

    // one big batched vocab projection: (4032 x 512) @ (512 x 10000)
    const int MT = MFC / 64, NT = (V_ + 63) / 64;
    dim3 grid(NT, (MT + 7) / 8);             // 157 x 8 blocks, 8 waves each
    k_fc<<<grid, 256, 0, stream>>>(Hall, Wfc_bf, bfc, clen, out);
}
